// Aggregator_80590766342883
// MI455X (gfx1250) — matrix operand order repaired and checked by reading, not compiled, not run on hardware
//
#include <hip/hip_runtime.h>
#include <hip/hip_bf16.h>

typedef __attribute__((ext_vector_type(16))) __bf16          v16bf;
typedef __attribute__((ext_vector_type(16))) unsigned short  v16us;
typedef __attribute__((ext_vector_type(8)))  unsigned short  v8us;
typedef __attribute__((ext_vector_type(8)))  unsigned int    v8ui;
typedef __attribute__((ext_vector_type(2)))  __bf16          v2bf;
typedef __attribute__((ext_vector_type(8)))  float           v8f;

#define KNB 20   // neighbors per row
#define DD  64   // embedding dim

static __device__ __forceinline__ unsigned short f2bf(float f) {
  // round-to-nearest-even f32 -> bf16 (software; used by staging and the f32 fallback)
  unsigned int u = __float_as_uint(f);
  u += 0x7fffu + ((u >> 16) & 1u);
  return (unsigned short)(u >> 16);
}

static __device__ __forceinline__ unsigned int pk2bf(float a, float b) {
#if __has_builtin(__builtin_amdgcn_cvt_pk_bf16_f32)
  v2bf t = __builtin_amdgcn_cvt_pk_bf16_f32(a, b);
  return __builtin_bit_cast(unsigned int, t);
#else
  return (unsigned int)f2bf(a) | ((unsigned int)f2bf(b) << 16);
#endif
}

#define SHUF16(lo, hi) \
  __builtin_shufflevector(lo, hi, 0,1,2,3,4,5,6,7,8,9,10,11,12,13,14,15)

// ---------------------------------------------------------------- zero fill
__global__ void zero_f32(float* __restrict__ p, int n) {
  int i = blockIdx.x * blockDim.x + threadIdx.x;
  if (i < n) p[i] = 0.0f;
}

// -------------------------------------------- one-shot bf16 staging kernels
__global__ void cvt_bf16_kernel(const float* __restrict__ src,
                                unsigned short* __restrict__ dst, int n) {
  int i = blockIdx.x * 256 + threadIdx.x;
  if (i < n) dst[i] = f2bf(src[i]);
}

__global__ void cvt_bf16T_kernel(const float* __restrict__ src,      // [n][64]
                                 unsigned short* __restrict__ dst,   // [n][64]
                                 unsigned short* __restrict__ dstT,  // [64][n]
                                 int n) {
  int i = blockIdx.x * 256 + threadIdx.x;
  if (i >= n * DD) return;
  int r = i >> 6, d = i & 63;
  unsigned short b = f2bf(src[i]);
  dst[i] = b;
  dstT[(long)d * n + r] = b;
}

// ------------------------------------------------ _agg (news & entity rows)
__global__ __launch_bounds__(256) void agg_kernel(
    const float* __restrict__ head, const int* __restrict__ ent_idx,
    const int* __restrict__ rel_idx, const float* __restrict__ ent_emb,
    const float* __restrict__ rel_emb, float* __restrict__ out, int n) {
  const int wid  = threadIdx.x >> 5;
  const int lane = threadIdx.x & 31;
  const int row  = blockIdx.x * 8 + wid;
  if (row >= n) return;

  const float h0 = head[row * DD + lane];
  const float h1 = head[row * DD + lane + 32];
  const int* ei = ent_idx + row * KNB;   // wave-uniform -> scalar loads
  const int* ri = rel_idx + row * KNB;

  float a0 = 0.f, a1 = 0.f;
  #pragma unroll
  for (int k = 0; k < KNB; ++k) {
    const float* rp = rel_emb + (long)ri[k] * DD;
    float t0 = h0 * rp[lane], t1 = h1 * rp[lane + 32];
    a0 += t0 * t0; a1 += t1 * t1;
  }
  const float hn0 = sqrtf(a0), hn1 = sqrtf(a1);

  float att[KNB];
  #pragma unroll
  for (int k = 0; k < KNB; ++k) {
    const float* tp = ent_emb + (long)ei[k] * DD;
    const float* rp = rel_emb + (long)ri[k] * DD;
    float v = tp[lane] * rp[lane] * hn0 + tp[lane + 32] * rp[lane + 32] * hn1;
    #pragma unroll
    for (int m = 1; m < 32; m <<= 1) v += __shfl_xor(v, m, 32);
    att[k] = v;
  }
  float mx = -1e30f;
  #pragma unroll
  for (int k = 0; k < KNB; ++k) { att[k] = att[k] * att[k]; mx = fmaxf(mx, att[k]); }
  float se = 0.f;
  #pragma unroll
  for (int k = 0; k < KNB; ++k) { att[k] = __expf(att[k] - mx); se += att[k]; }
  const float inv = 1.0f / se;

  float o0 = 0.f, o1 = 0.f;
  #pragma unroll
  for (int k = 0; k < KNB; ++k) {
    const float w = att[k] * inv;
    const float* tp = ent_emb + (long)ei[k] * DD;
    o0 += w * tp[lane]; o1 += w * tp[lane + 32];
  }
  out[row * DD + lane]      = o0;
  out[row * DD + lane + 32] = o1;
}

// ------------------------------------------- sparse segment-sum into users
__global__ __launch_bounds__(256) void scatter_kernel(
    const int* __restrict__ rows, const int* __restrict__ cols,
    const float* __restrict__ vals, const float* __restrict__ news_agg,
    float* __restrict__ base, int nnz) {
  int g = blockIdx.x * 256 + threadIdx.x;
  int i = g >> 5, lane = g & 31;
  if (i >= nnz) return;
  int r = rows[i], c = cols[i];
  float v = vals[i];
  atomicAdd(&base[(long)r * DD + lane],      v * news_agg[(long)c * DD + lane]);
  atomicAdd(&base[(long)r * DD + lane + 32], v * news_agg[(long)c * DD + lane + 32]);
}

// ===================== fused flash attention, bf16-staged fast path ========
// out = base*(1 + softmax(U Vᵀ) V).  Block = 16 users, 8 waves split news,
// each iteration consumes a PAIR of 16-news tiles so the P·V WMMA runs with
// a fully populated K=32 (no zero padding).
__global__ __launch_bounds__(256) void attn_kernel_bf16(
    const unsigned short* __restrict__ ubf,   // [NU][64] bf16
    const unsigned short* __restrict__ vbf,   // [NN][64] bf16
    const unsigned short* __restrict__ vbfT,  // [64][NN] bf16
    float* __restrict__ user_io,              // [NU,64] in: base, out: final
    int n_news) {
  __shared__ float sm[8][16];
  __shared__ float ss[8][16];
  __shared__ float sO[8][16][64];

  const int wid  = threadIdx.x >> 5;
  const int lane = threadIdx.x & 31;
  const int h    = lane >> 4;
  const int q    = lane & 15;
  const int u0   = blockIdx.x * 16;

  // B operand (Uᵀ): lane = user column q, element e <-> K(dim) = 8h + e (e<8), 16 + 8h + (e-8) (e>=8) (+32)
  const unsigned short* up = ubf + (long)(u0 + q) * DD + 8 * h;
  v8us u00 = *(const v8us*)(up);
  v8us u01 = *(const v8us*)(up + 16);
  v8us u10 = *(const v8us*)(up + 32);
  v8us u11 = *(const v8us*)(up + 48);
  const v16bf ub0 = __builtin_bit_cast(v16bf, SHUF16(u00, u01));
  const v16bf ub1 = __builtin_bit_cast(v16bf, SHUF16(u10, u11));

  float m_run = -1e30f, s_run = 0.f;
  v8f oacc[4] = {};
  const v8us z8 = {};

  const int ntiles = n_news >> 4;
  const int npairs = ntiles >> 1;

  for (int tp = wid; tp < npairs; tp += 8) {
    const int nb = tp << 5;   // 32 news per iteration

    // ---- scores for both tiles of the pair: Sᵀ = V·Uᵀ
    v8f sa = {}, sb = {};
    {
      const unsigned short* vp = vbf + (long)(nb + q) * DD + 8 * h;
      v8us a0 = *(const v8us*)(vp);
      v8us a1 = *(const v8us*)(vp + 16);
      v8us a2 = *(const v8us*)(vp + 32);
      v8us a3 = *(const v8us*)(vp + 48);
      sa = __builtin_amdgcn_wmma_f32_16x16x32_bf16(false, __builtin_bit_cast(v16bf, SHUF16(a0, a1)),
                                                   false, ub0, (short)0, sa, false, false);
      sa = __builtin_amdgcn_wmma_f32_16x16x32_bf16(false, __builtin_bit_cast(v16bf, SHUF16(a2, a3)),
                                                   false, ub1, (short)0, sa, false, false);
      const unsigned short* vp2 = vbf + (long)(nb + 16 + q) * DD + 8 * h;
      v8us b0 = *(const v8us*)(vp2);
      v8us b1 = *(const v8us*)(vp2 + 16);
      v8us b2 = *(const v8us*)(vp2 + 32);
      v8us b3 = *(const v8us*)(vp2 + 48);
      sb = __builtin_amdgcn_wmma_f32_16x16x32_bf16(false, __builtin_bit_cast(v16bf, SHUF16(b0, b1)),
                                                   false, ub0, (short)0, sb, false, false);
      sb = __builtin_amdgcn_wmma_f32_16x16x32_bf16(false, __builtin_bit_cast(v16bf, SHUF16(b2, b3)),
                                                   false, ub1, (short)0, sb, false, false);
    }

    // ---- online softmax over this user's 32 tile scores
    float pm = sa[0];
    #pragma unroll
    for (int r = 1; r < 8; ++r) pm = fmaxf(pm, sa[r]);
    #pragma unroll
    for (int r = 0; r < 8; ++r) pm = fmaxf(pm, sb[r]);
    pm = fmaxf(pm, __shfl_xor(pm, 16, 32));
    const float m_new = fmaxf(m_run, pm);
    const float alpha = __expf(m_run - m_new);
    float pA[8], pB[8]; float ps = 0.f;
    #pragma unroll
    for (int r = 0; r < 8; ++r) {
      pA[r] = __expf(sa[r] - m_new);
      pB[r] = __expf(sb[r] - m_new);
      ps += pA[r] + pB[r];
    }
    ps += __shfl_xor(ps, 16, 32);
    s_run = s_run * alpha + ps;
    m_run = m_new;
    #pragma unroll
    for (int c = 0; c < 4; ++c) oacc[c] *= alpha;

    // ---- B operand (Pᵀ, 32x16 dense):
    // half 0 holds K rows 0..15 (tile A), half 1 holds K rows 16..31 (tile B).
    // Each lane sends the array its partner half needs.
    float o[8];
    #pragma unroll
    for (int r = 0; r < 8; ++r) o[r] = __shfl_xor(h ? pA[r] : pB[r], 16, 32);
    v8ui pbi;
    #pragma unroll
    for (int j = 0; j < 4; ++j) {
      float l0 = h ? o[2*j]      : pA[2*j];
      float l1 = h ? o[2*j + 1]  : pA[2*j + 1];
      float g0 = h ? pB[2*j]     : o[2*j];
      float g1 = h ? pB[2*j + 1] : o[2*j + 1];
      pbi[j]     = pk2bf(l0, l1);
      pbi[4 + j] = pk2bf(g0, g1);
    }
    const v16bf pbv = __builtin_bit_cast(v16bf, pbi);

    // ---- A operand (Vᵀ, K = 32 news): two contiguous b128s per 16-dim chunk
    #pragma unroll
    for (int c = 0; c < 4; ++c) {
      const unsigned short* tb = vbfT + (long)(16 * c + q) * n_news + nb + 16 * h;
      v8us tlo = *(const v8us*)(tb);        // K = 16h + 0..7
      v8us thi = *(const v8us*)(tb + 8);    // K = 16h + 8..15
      oacc[c] = __builtin_amdgcn_wmma_f32_16x16x32_bf16(
          false, __builtin_bit_cast(v16bf, SHUF16(tlo, thi)),
          false, pbv, (short)0, oacc[c], false, false);
    }
  }

  // ---- odd remainder tile (single 16-news, half-populated P·V)
  if ((ntiles & 1) && wid == (npairs & 7)) {
    const int nb = (ntiles - 1) << 4;
    const unsigned short* vp = vbf + (long)(nb + q) * DD + 8 * h;
    v8us a0 = *(const v8us*)(vp);
    v8us a1 = *(const v8us*)(vp + 16);
    v8us a2 = *(const v8us*)(vp + 32);
    v8us a3 = *(const v8us*)(vp + 48);
    v8f s = {};
    s = __builtin_amdgcn_wmma_f32_16x16x32_bf16(false, __builtin_bit_cast(v16bf, SHUF16(a0, a1)),
                                                false, ub0, (short)0, s, false, false);
    s = __builtin_amdgcn_wmma_f32_16x16x32_bf16(false, __builtin_bit_cast(v16bf, SHUF16(a2, a3)),
                                                false, ub1, (short)0, s, false, false);
    float pm = s[0];
    #pragma unroll
    for (int r = 1; r < 8; ++r) pm = fmaxf(pm, s[r]);
    pm = fmaxf(pm, __shfl_xor(pm, 16, 32));
    const float m_new = fmaxf(m_run, pm);
    const float alpha = __expf(m_run - m_new);
    float p[8]; float ps = 0.f;
    #pragma unroll
    for (int r = 0; r < 8; ++r) { p[r] = __expf(s[r] - m_new); ps += p[r]; }
    ps += __shfl_xor(ps, 16, 32);
    s_run = s_run * alpha + ps;
    m_run = m_new;
    #pragma unroll
    for (int c = 0; c < 4; ++c) oacc[c] *= alpha;

    float po[8];
    #pragma unroll
    for (int r = 0; r < 8; ++r) po[r] = __shfl_xor(p[r], 16, 32);
    v8ui pbi;
    #pragma unroll
    for (int j = 0; j < 4; ++j) pbi[j]     = pk2bf(p[2*j],  p[2*j+1]);
    #pragma unroll
    for (int j = 0; j < 4; ++j) pbi[4 + j] = 0u;
    const v16bf pbv = __builtin_bit_cast(v16bf, pbi);

    #pragma unroll
    for (int c = 0; c < 4; ++c) {
      v8us tlo = *(const v8us*)(vbfT + (long)(16 * c + q) * n_news + nb + 8 * h);
      oacc[c] = __builtin_amdgcn_wmma_f32_16x16x32_bf16(
          false, __builtin_bit_cast(v16bf, SHUF16(tlo, z8)),
          false, pbv, (short)0, oacc[c], false, false);
    }
  }

  // ---- cross-wave flash combine through LDS
  if (h == 0) { sm[wid][q] = m_run; ss[wid][q] = s_run; }
  #pragma unroll
  for (int c = 0; c < 4; ++c)
    #pragma unroll
    for (int r = 0; r < 8; ++r)
      sO[wid][q][c * 16 + r + 8 * h] = oacc[c][r];
  __syncthreads();

  #pragma unroll
  for (int i = 0; i < 4; ++i) {
    int idx = threadIdx.x + 256 * i;
    int u = idx >> 6, d = idx & 63;
    float gm = -1e30f;
    #pragma unroll
    for (int w = 0; w < 8; ++w) gm = fmaxf(gm, sm[w][u]);
    float S = 0.f, val = 0.f;
    #pragma unroll
    for (int w = 0; w < 8; ++w) {
      float e = __expf(sm[w][u] - gm);
      S   += ss[w][u] * e;
      val += sO[w][u][d] * e;
    }
    const float attn = val / S;
    float* op = user_io + (long)(u0 + u) * DD + d;
    const float b = *op;
    *op = b + attn * b;
  }
}

// ===================== fallback: on-the-fly conversion (ws too small) ======
__global__ __launch_bounds__(256) void attn_kernel_f32(
    const float* __restrict__ user_emb, const float* __restrict__ news_agg,
    float* __restrict__ user_io, int n_news) {
  __shared__ float sm[8][16];
  __shared__ float ss[8][16];
  __shared__ float sO[8][16][64];

  const int wid  = threadIdx.x >> 5;
  const int lane = threadIdx.x & 31;
  const int h    = lane >> 4;
  const int q    = lane & 15;
  const int u0   = blockIdx.x * 16;

  v16us ubu0, ubu1;
  {
    const float* up = user_emb + (long)(u0 + q) * DD + 8 * h;
    #pragma unroll
    for (int e = 0; e < 16; ++e) { ubu0[e] = f2bf(up[e + (e & 8)]); ubu1[e] = f2bf(up[e + (e & 8) + 32]); }
  }
  const v16bf ub0 = __builtin_bit_cast(v16bf, ubu0);
  const v16bf ub1 = __builtin_bit_cast(v16bf, ubu1);

  float m_run = -1e30f, s_run = 0.f;
  v8f oacc[4] = {};

  const int ntiles = n_news >> 4;
  for (int t = wid; t < ntiles; t += 8) {
    const int nb = t << 4;
    v16us va0, va1;
    const float* vp = news_agg + (long)(nb + q) * DD + 8 * h;
    #pragma unroll
    for (int e = 0; e < 8; ++e) {
      va0[e]     = f2bf(vp[e]);
      va0[e + 8] = f2bf(vp[e + 16]);
      va1[e]     = f2bf(vp[e + 32]);
      va1[e + 8] = f2bf(vp[e + 48]);
    }
    v8f s = {};
    s = __builtin_amdgcn_wmma_f32_16x16x32_bf16(false, __builtin_bit_cast(v16bf, va0),
                                                false, ub0, (short)0, s, false, false);
    s = __builtin_amdgcn_wmma_f32_16x16x32_bf16(false, __builtin_bit_cast(v16bf, va1),
                                                false, ub1, (short)0, s, false, false);

    float pm = s[0];
    #pragma unroll
    for (int r = 1; r < 8; ++r) pm = fmaxf(pm, s[r]);
    pm = fmaxf(pm, __shfl_xor(pm, 16, 32));
    const float m_new = fmaxf(m_run, pm);
    const float alpha = __expf(m_run - m_new);
    float p[8]; float ps = 0.f;
    #pragma unroll
    for (int r = 0; r < 8; ++r) { p[r] = __expf(s[r] - m_new); ps += p[r]; }
    ps += __shfl_xor(ps, 16, 32);
    s_run = s_run * alpha + ps;
    m_run = m_new;
    #pragma unroll
    for (int c = 0; c < 4; ++c) oacc[c] *= alpha;

    float po[8];
    #pragma unroll
    for (int r = 0; r < 8; ++r) po[r] = __shfl_xor(p[r], 16, 32);
    v8ui pbi;
    #pragma unroll
    for (int j = 0; j < 4; ++j) pbi[j]     = pk2bf(p[2*j],  p[2*j+1]);
    #pragma unroll
    for (int j = 0; j < 4; ++j) pbi[4 + j] = 0u;
    const v16bf pbv = __builtin_bit_cast(v16bf, pbi);

    const float* vb = news_agg + (long)nb * DD;
    #pragma unroll
    for (int c = 0; c < 4; ++c) {
      v16us av = {};
      #pragma unroll
      for (int e = 0; e < 8; ++e)
        av[e] = f2bf(vb[(e + 8 * h) * DD + c * 16 + q]);
      oacc[c] = __builtin_amdgcn_wmma_f32_16x16x32_bf16(false, __builtin_bit_cast(v16bf, av),
                                                        false, pbv, (short)0, oacc[c], false, false);
    }
  }

  if (h == 0) { sm[wid][q] = m_run; ss[wid][q] = s_run; }
  #pragma unroll
  for (int c = 0; c < 4; ++c)
    #pragma unroll
    for (int r = 0; r < 8; ++r)
      sO[wid][q][c * 16 + r + 8 * h] = oacc[c][r];
  __syncthreads();

  #pragma unroll
  for (int i = 0; i < 4; ++i) {
    int idx = threadIdx.x + 256 * i;
    int u = idx >> 6, d = idx & 63;
    float gm = -1e30f;
    #pragma unroll
    for (int w = 0; w < 8; ++w) gm = fmaxf(gm, sm[w][u]);
    float S = 0.f, val = 0.f;
    #pragma unroll
    for (int w = 0; w < 8; ++w) {
      float e = __expf(sm[w][u] - gm);
      S   += ss[w][u] * e;
      val += sO[w][u][d] * e;
    }
    const float attn = val / S;
    float* op = user_io + (long)(u0 + u) * DD + d;
    const float b = *op;
    *op = b + attn * b;
  }
}

extern "C" void kernel_launch(void* const* d_in, const int* in_sizes, int n_in,
                              void* d_out, int out_size, void* d_ws, size_t ws_size,
                              hipStream_t stream) {
  const float* user_emb     = (const float*)d_in[0];
  const float* news_emb     = (const float*)d_in[1];
  const float* entity_emb   = (const float*)d_in[2];
  const float* relation_emb = (const float*)d_in[3];
  const int* news_entities  = (const int*)d_in[4];
  const int* news_relations = (const int*)d_in[5];
  const int* neigh_entities = (const int*)d_in[6];
  const int* neigh_relations= (const int*)d_in[7];
  const int* irows          = (const int*)d_in[8];
  const int* icols          = (const int*)d_in[9];
  const float* ivals        = (const float*)d_in[10];

  const int NU  = in_sizes[0] / DD;   // 10000
  const int NN  = in_sizes[1] / DD;   // 10000
  const int NE  = in_sizes[2] / DD;   // 100000
  const int NNZ = in_sizes[8];        // 500000

  float* out        = (float*)d_out;
  float* news_agg   = out;                              // [NN*64]
  float* entity_agg = out + (long)NN * DD;              // [NE*64]
  float* user_agg   = out + (long)(NN + NE) * DD;       // [NU*64]

  // 1) zero user accumulator
  zero_f32<<<(NU * DD + 255) / 256, 256, 0, stream>>>(user_agg, NU * DD);
  // 2) news aggregation
  agg_kernel<<<(NN + 7) / 8, 256, 0, stream>>>(news_emb, news_entities, news_relations,
                                               entity_emb, relation_emb, news_agg, NN);
  // 3) entity aggregation
  agg_kernel<<<(NE + 7) / 8, 256, 0, stream>>>(entity_emb, neigh_entities, neigh_relations,
                                               entity_emb, relation_emb, entity_agg, NE);
  // 4) sparse interact_mat @ news_agg -> user base
  scatter_kernel<<<((long)NNZ * 32 + 255) / 256, 256, 0, stream>>>(
      irows, icols, ivals, news_agg, user_agg, NNZ);

  // 5) attention: bf16-staged fast path if workspace fits, else on-the-fly
  const size_t needWS = ((size_t)NU * DD + 2 * (size_t)NN * DD) * sizeof(unsigned short);
  if (ws_size >= needWS) {
    unsigned short* ubf  = (unsigned short*)d_ws;                 // [NU][64]
    unsigned short* vbf  = ubf + (size_t)NU * DD;                 // [NN][64]
    unsigned short* vbfT = vbf + (size_t)NN * DD;                 // [64][NN]
    cvt_bf16_kernel<<<(NU * DD + 255) / 256, 256, 0, stream>>>(user_emb, ubf, NU * DD);
    cvt_bf16T_kernel<<<(NN * DD + 255) / 256, 256, 0, stream>>>(news_agg, vbf, vbfT, NN);
    attn_kernel_bf16<<<NU / 16, 256, 0, stream>>>(ubf, vbf, vbfT, user_agg, NN);
  } else {
    attn_kernel_f32<<<NU / 16, 256, 0, stream>>>(user_emb, news_agg, user_agg, NN);
  }
}
